// VersatileConvSE3_68547678044849
// MI455X (gfx1250) — matrix operand order repaired and checked by reading, not compiled, not run on hardware
//
#include <hip/hip_runtime.h>
#include <hip/hip_bf16.h>

typedef __attribute__((ext_vector_type(16))) __bf16 v16bf;
typedef __attribute__((ext_vector_type(8)))  __bf16 v8bf;
typedef __attribute__((ext_vector_type(8)))  float  v8f;

#define E_TOTAL 16384
#define PASS_E  2048
#define N_PASS  8

// K index for 16-bit A/B operand slot j in lane-half hi:
// j<8 -> 8*hi + j ; j>=8 -> 16 + 8*hi + (j-8)
__device__ __forceinline__ int kslot(int j, int hi) {
    return (j & 7) + 8 * (((j >> 3) << 1) + hi);
}

// Async copy of 32 bytes/lane from global (saddr base + per-lane byte offset)
// into LDS at per-lane byte address. INST_OFFSET applies to both addresses.
__device__ __forceinline__ void async_tile_load(const __bf16* base,
                                                unsigned ldsaddr, unsigned goff) {
    asm volatile("global_load_async_to_lds_b128 %0, %1, %2\n\t"
                 "global_load_async_to_lds_b128 %0, %1, %2 offset:16"
                 :: "v"(ldsaddr), "v"(goff), "s"(base)
                 : "memory");
}

// ---------------------------------------------------------------------------
// K0: repack w3 [45056 x 32] fp32 -> bf16 WMMA-A tiles [2816 tiles x 512]
// tile t covers rows k' = t*16 .. t*16+15, K = m (32). One wave per tile.
// ---------------------------------------------------------------------------
__global__ __launch_bounds__(256) void k0_pack_w3(const float* __restrict__ w3,
                                                  __bf16* __restrict__ w3p) {
    const int wv = threadIdx.x >> 5, l = threadIdx.x & 31;
    const int hi = l >> 4, ln = l & 15;
    const int tile = blockIdx.x * 8 + wv;            // 352*8 = 2816 exactly
    const float* src = w3 + ((size_t)tile * 16 + ln) * 32;
    v8bf lo, hh;
#pragma unroll
    for (int j = 0; j < 8; ++j) {
        lo[j] = (__bf16)src[8 * hi + j];             // K = 0..7 / 8..15
        hh[j] = (__bf16)src[16 + 8 * hi + j];        // K = 16..23 / 24..31
    }
    __bf16* d = w3p + (size_t)tile * 512 + l * 16;
    *(v8bf*)d       = lo;
    *(v8bf*)(d + 8) = hh;
}

// ---------------------------------------------------------------------------
// K1: radial MLP first two layers: x[16] -> h[32] (Linear+LN+ReLU x2), bf16 out
// One thread per edge.
// ---------------------------------------------------------------------------
__global__ __launch_bounds__(256) void k1_mlp(const float* __restrict__ x,
        const float* __restrict__ w1, const float* __restrict__ b1,
        const float* __restrict__ g1, const float* __restrict__ be1,
        const float* __restrict__ w2, const float* __restrict__ b2,
        const float* __restrict__ g2, const float* __restrict__ be2,
        __bf16* __restrict__ hbf) {
    __shared__ float sW1[512];
    __shared__ float sW2[1024];
    __shared__ float sP[192];
    for (int i = threadIdx.x; i < 512; i += 256) sW1[i] = w1[i];
    for (int i = threadIdx.x; i < 1024; i += 256) sW2[i] = w2[i];
    if (threadIdx.x < 32) {
        int t = threadIdx.x;
        sP[t] = b1[t]; sP[32 + t] = g1[t]; sP[64 + t] = be1[t];
        sP[96 + t] = b2[t]; sP[128 + t] = g2[t]; sP[160 + t] = be2[t];
    }
    __syncthreads();
    const int e = blockIdx.x * 256 + threadIdx.x;
    float xv[16];
#pragma unroll
    for (int i = 0; i < 16; ++i) xv[i] = x[(size_t)e * 16 + i];
    float t1[32], t2[32];
    float mu = 0.f;
#pragma unroll
    for (int o = 0; o < 32; ++o) {
        float a = sP[o];
#pragma unroll
        for (int i = 0; i < 16; ++i) a += sW1[o * 16 + i] * xv[i];
        t1[o] = a; mu += a;
    }
    mu *= (1.f / 32.f);
    float var = 0.f;
#pragma unroll
    for (int o = 0; o < 32; ++o) { float d = t1[o] - mu; var += d * d; }
    float is = rsqrtf(var * (1.f / 32.f) + 1e-5f);
#pragma unroll
    for (int o = 0; o < 32; ++o)
        t1[o] = fmaxf(0.f, (t1[o] - mu) * is * sP[32 + o] + sP[64 + o]);
    mu = 0.f;
#pragma unroll
    for (int o = 0; o < 32; ++o) {
        float a = sP[96 + o];
#pragma unroll
        for (int m = 0; m < 32; ++m) a += sW2[o * 32 + m] * t1[m];
        t2[o] = a; mu += a;
    }
    mu *= (1.f / 32.f);
    var = 0.f;
#pragma unroll
    for (int o = 0; o < 32; ++o) { float d = t2[o] - mu; var += d * d; }
    is = rsqrtf(var * (1.f / 32.f) + 1e-5f);
#pragma unroll
    for (int o = 0; o < 32; ++o) {
        float h = fmaxf(0.f, (t2[o] - mu) * is * sP[128 + o] + sP[160 + o]);
        hbf[(size_t)e * 32 + o] = (__bf16)h;
    }
}

// ---------------------------------------------------------------------------
// K2: tmp[e][k=(c*44+f)][s] = sum_i feat[e][c][i]*basis[e][i][f][s], bf16,
// written as WMMA-B packed tiles of 32k x 16s (1 KB each), 44 tiles/edge.
// One block (8 waves) per edge. WMMA with K=16 zero-padded to 32.
// ---------------------------------------------------------------------------
__global__ __launch_bounds__(256) void k2_tmp(const float* __restrict__ feat,
        const float* __restrict__ basis, __bf16* __restrict__ tmpp, int eBase) {
    __shared__ __align__(16) __bf16 TmpE[1408 * 16];   // 45 KB
    const int e = eBase + blockIdx.x;
    const int tid = threadIdx.x, wv = tid >> 5, l = tid & 31;
    const int hi = l >> 4, ln = l & 15;
    const int cb = wv & 1;                              // c-block of this wave
    // A = feat tile [M=c 16][K=i, 16 real + 16 zero-pad]
    v16bf A;
    {
        const float* fp = feat + ((size_t)e * 32 + cb * 16 + ln) * 16 + 8 * hi;
#pragma unroll
        for (int j = 0; j < 8; ++j) { A[j] = (__bf16)fp[j]; A[j + 8] = (__bf16)0.0f; }
    }
    for (int s = 0; s < 11; ++s) {
        const int nt = (wv >> 1) + 4 * s;               // f index 0..43
        // B = basis tile [K=i][N=s], i = 8*hi + j (j<8), zero-padded above 16
        v16bf B;
        const float* bp = basis + (((size_t)e * 16 + 8 * hi) * 44 + nt) * 16 + ln;
#pragma unroll
        for (int j = 0; j < 8; ++j) { B[j] = (__bf16)bp[(size_t)j * 704]; B[j + 8] = (__bf16)0.0f; }
        v8f z = {};
        v8f d = __builtin_amdgcn_wmma_f32_16x16x32_bf16(false, A, false, B,
                                                        (short)0, z, false, false);
#pragma unroll
        for (int v = 0; v < 8; ++v) {
            const int c = cb * 16 + v + 8 * hi;
            TmpE[(c * 44 + nt) * 16 + ln] = (__bf16)d[v];
        }
    }
    __syncthreads();
    // Repack LDS [k][s] -> global packed B-operand tiles (lane image)
    unsigned int* dst = (unsigned int*)(tmpp + (size_t)blockIdx.x * 44 * 512);
    for (int idx = tid; idx < 44 * 256; idx += 256) {
        const int tile = idx >> 8, word = idx & 255;
        const int ld = word >> 3, j2 = word & 7;        // dest lane, u32 slot
        const int lhi = ld >> 4, lln = ld & 15;
        const int r = kslot(2 * j2, lhi);               // K row pair (r, r+1)
        const int k = tile * 32 + r;
        const unsigned int a = __builtin_bit_cast(unsigned short, TmpE[k * 16 + lln]);
        const unsigned int b = __builtin_bit_cast(unsigned short, TmpE[(k + 1) * 16 + lln]);
        dst[idx] = a | (b << 16);
    }
}

// ---------------------------------------------------------------------------
// K3: fused radial generation + consumption. 16 waves, 16 edges per block.
// Loop kc (44 chunks of 32 k):
//   async:   next chunk's tmp tiles -> LDS (double buffered, ASYNCcnt)
//   phase B: 64 wmma tiles R[k'16 x e16] = w3_tile @ H   -> LDS (bf16)
//   phase C: per-edge wave: out[o16 x s16] += R[o x k32] @ tmp[k32 x s16] (x2 o-blk)
// ---------------------------------------------------------------------------
__global__ __launch_bounds__(512) void k3_fused(const __bf16* __restrict__ w3p,
        const __bf16* __restrict__ hbf, const __bf16* __restrict__ tmpp,
        float* __restrict__ out, int eBase) {
    __shared__ __align__(16) __bf16 Rbuf[16 * 32 * 32];   // [e][o 0..31][k 0..31], 32 KB
    __shared__ __align__(16) __bf16 Tdb[2][16 * 512];     // tmp tiles, 2 x 16 KB
    const int tid = threadIdx.x;
    const int wv = tid >> 5, l = tid & 31;
    const int hi = l >> 4, ln = l & 15;

    // H B-operand [K=m 32][N=e 16], constant for the whole block
    v16bf hB;
    {
        const __bf16* hp = hbf + (size_t)(eBase + blockIdx.x * 16 + ln) * 32;
        v8bf lo = *(const v8bf*)(hp + 8 * hi);
        v8bf hh = *(const v8bf*)(hp + 16 + 8 * hi);
#pragma unroll
        for (int j = 0; j < 8; ++j) { hB[j] = lo[j]; hB[j + 8] = hh[j]; }
    }
    v8f acc0 = {}, acc1 = {};
    // per-wave edge: byte base of this edge's 44 packed tiles in tmpp
    const unsigned tmpEdgeByte = (unsigned)(blockIdx.x * 16 + wv) * 44u * 1024u + (unsigned)(l * 32);
    // LDS byte addresses (low 32 bits of generic pointer = LDS offset)
    const unsigned ldsT = (unsigned)(size_t)(void*)&Tdb[0][0] + (unsigned)(wv * 1024 + l * 32);

    // kick off chunk 0 into buffer 0 (each wave loads only its own edge tile)
    async_tile_load(tmpp, ldsT, tmpEdgeByte);

    for (int kc = 0; kc < 44; ++kc) {
        const int buf = kc & 1;
        if (kc < 43)   // prefetch next chunk's tile into the other buffer
            async_tile_load(tmpp, ldsT + (buf ^ 1) * 16384u,
                            tmpEdgeByte + (unsigned)(kc + 1) * 1024u);
        // ---- phase B: this wave computes 4 radial tiles ----
#pragma unroll
        for (int q = 0; q < 4; ++q) {
            const int t = wv * 4 + q;                 // 0..63
            const int o = t >> 1;                     // 0..31 (includes o-block)
            const int h = t & 1;                      // k half within chunk
            const int w3tile = o * 88 + kc * 2 + h;   // k'/16
            const __bf16* ap = w3p + (size_t)w3tile * 512 + l * 16;
            if (kc < 43) __builtin_prefetch(ap + 1024, 0, 0);  // next kc, same slot
            v16bf Aw = *(const v16bf*)ap;
            v8f z = {};
            v8f d = __builtin_amdgcn_wmma_f32_16x16x32_bf16(false, Aw, false, hB,
                                                            (short)0, z, false, false);
            v8bf rb;
#pragma unroll
            for (int v = 0; v < 8; ++v) rb[v] = (__bf16)d[v];
            // rows k_local = h*16 + 8*hi + v (contiguous in v) -> one b128 store
            *(v8bf*)(&Rbuf[ln * 1024 + o * 32 + h * 16 + 8 * hi]) = rb;
        }
        // current buffer's async loads are the oldest outstanding ones:
        // allow the 2 newest (next chunk) to remain in flight.
        if (kc < 43) asm volatile("s_wait_asynccnt 2" ::: "memory");
        else         asm volatile("s_wait_asynccnt 0" ::: "memory");
        __syncthreads();
        // ---- phase C: wave wv consumes for edge wv ----
        {
            const __bf16* bp = &Tdb[buf][wv * 512 + l * 16];
            v16bf B = *(const v16bf*)bp;              // packed tmp tile from LDS
            const __bf16* r0 = &Rbuf[wv * 1024 + ln * 32];          // o-block 0
            const __bf16* r1 = &Rbuf[wv * 1024 + (16 + ln) * 32];   // o-block 1
            v8bf a0l = *(const v8bf*)(r0 + 8 * hi);
            v8bf a0h = *(const v8bf*)(r0 + 16 + 8 * hi);
            v8bf a1l = *(const v8bf*)(r1 + 8 * hi);
            v8bf a1h = *(const v8bf*)(r1 + 16 + 8 * hi);
            v16bf A0, A1;
#pragma unroll
            for (int j = 0; j < 8; ++j) {
                A0[j] = a0l[j]; A0[j + 8] = a0h[j];
                A1[j] = a1l[j]; A1[j + 8] = a1h[j];
            }
            acc0 = __builtin_amdgcn_wmma_f32_16x16x32_bf16(false, A0, false, B,
                                                           (short)0, acc0, false, false);
            acc1 = __builtin_amdgcn_wmma_f32_16x16x32_bf16(false, A1, false, B,
                                                           (short)0, acc1, false, false);
        }
        __syncthreads();
    }
    // ---- store out[e][o][s] fp32 ----
    const size_t eo = (size_t)(eBase + blockIdx.x * 16 + wv) * 512;
#pragma unroll
    for (int v = 0; v < 8; ++v) {
        out[eo + (v + 8 * hi) * 16 + ln]        = acc0[v];
        out[eo + (16 + v + 8 * hi) * 16 + ln]   = acc1[v];
    }
}

// ---------------------------------------------------------------------------
extern "C" void kernel_launch(void* const* d_in, const int* in_sizes, int n_in,
                              void* d_out, int out_size, void* d_ws, size_t ws_size,
                              hipStream_t stream) {
    (void)in_sizes; (void)n_in; (void)out_size; (void)ws_size;
    const float* feat  = (const float*)d_in[0];
    const float* ief   = (const float*)d_in[1];
    const float* basis = (const float*)d_in[2];
    const float* w1  = (const float*)d_in[3];
    const float* b1  = (const float*)d_in[4];
    const float* g1  = (const float*)d_in[5];
    const float* be1 = (const float*)d_in[6];
    const float* w2  = (const float*)d_in[7];
    const float* b2  = (const float*)d_in[8];
    const float* g2  = (const float*)d_in[9];
    const float* be2 = (const float*)d_in[10];
    const float* w3  = (const float*)d_in[11];
    float* out = (float*)d_out;

    char* ws = (char*)d_ws;
    __bf16* w3p  = (__bf16*)ws;                         // 2816*1024 B = 2,883,584
    __bf16* hbf  = (__bf16*)(ws + 2884608);             // 16384*64 B = 1,048,576
    __bf16* tmpp = (__bf16*)(ws + 2884608 + 1048576);   // 2048*44*1024 B = 92.3 MB

    k0_pack_w3<<<352, 256, 0, stream>>>(w3, w3p);
    k1_mlp<<<64, 256, 0, stream>>>(ief, w1, b1, g1, be1, w2, b2, g2, be2, hbf);
    for (int p = 0; p < N_PASS; ++p) {
        const int eBase = p * PASS_E;
        k2_tmp<<<PASS_E, 256, 0, stream>>>(feat, basis, tmpp, eBase);
        k3_fused<<<PASS_E / 16, 512, 0, stream>>>(w3p, hbf, tmpp, out, eBase);
    }
}
